// TensorProductModel_62457414419293
// MI455X (gfx1250) — matrix operand order repaired and checked by reading, not compiled, not run on hardware
//
#include <hip/hip_runtime.h>
#include <cmath>

// ---------------- constants ----------------
#define NN_NODES 512
#define NN_EDGES 8192
#define NN_GRAPHS 8
#define NN_FIN 64
#define NN_EMB 32
#define NN_MLP 256
#define RMAXF 10.0f

typedef __attribute__((ext_vector_type(8)))  float  v8f;
typedef __attribute__((ext_vector_type(16))) __bf16 v16bf;

struct LayerCfg {
  int numel, in_dim, out_dim, n_paths, n_out_fields;
  int out_off[3], out_l[3];
  int p_i1off[11], p_l1[11], p_shoff[11], p_l2[11];
  int p_outoff[11], p_l3[11], p_cg[11], p_coloff[11];
  float p_pw[11];
};

// ---------------- small helpers ----------------
__device__ __forceinline__ unsigned short f2bf(float f) {
  unsigned int u = __float_as_uint(f);
  unsigned int r = u + 0x7fffu + ((u >> 16) & 1u);
  return (unsigned short)(r >> 16);
}

// ---------------- CG / wigner-3j init (runs once, 11 threads) ----------------
__device__ double dfact(int n) { double r = 1.0; for (int i = 2; i <= n; ++i) r *= (double)i; return r; }

__device__ double su2cg(int j1, int m1, int j2, int m2, int j3, int m3) {
  if (m3 != m1 + m2) return 0.0;
  int vmin = max(max(-j1 + j2 + m3, -j1 + m1), 0);
  int vmax = min(min(j2 + j3 + m1, j3 - j1 + j2), j3 + m3);
  double C = sqrt((double)(2 * j3 + 1) * dfact(j3 + j1 - j2) * dfact(j3 - j1 + j2) *
                  dfact(j1 + j2 - j3) * dfact(j3 + m3) * dfact(j3 - m3) /
                  (dfact(j1 + j2 + j3 + 1) * dfact(j1 - m1) * dfact(j1 + m1) *
                   dfact(j2 - m2) * dfact(j2 + m2)));
  double S = 0.0;
  for (int v = vmin; v <= vmax; ++v) {
    double sgn = ((v + j2 + m2) & 1) ? -1.0 : 1.0;
    S += sgn * dfact(j2 + j3 + m1 - v) * dfact(j1 - m1 + v) /
         (dfact(v) * dfact(j3 - j1 + j2 - v) * dfact(j3 + m3 - v) * dfact(v + j1 - j2 - m3));
  }
  return C * S;
}

__device__ void build_q(int l, double qr[5][5], double qi[5][5]) {
  for (int i = 0; i < 5; ++i) for (int j = 0; j < 5; ++j) { qr[i][j] = 0.0; qi[i][j] = 0.0; }
  double inv = 1.0 / sqrt(2.0);
  for (int m = -l; m < 0; ++m) { qr[l + m][l - m] = inv; qi[l + m][l + m] = -inv; }
  qr[l][l] = 1.0;
  for (int m = 1; m <= l; ++m) {
    double sg = (m & 1) ? -1.0 : 1.0;
    qr[l + m][l + m] = sg * inv;
    qi[l + m][l - m] = sg * inv;
  }
  // multiply by (-i)^l
  double pr, pi;
  switch (l & 3) { case 0: pr = 1; pi = 0; break; case 1: pr = 0; pi = -1; break;
                   case 2: pr = -1; pi = 0; break; default: pr = 0; pi = 1; break; }
  for (int i = 0; i < 5; ++i) for (int j = 0; j < 5; ++j) {
    double a = qr[i][j], b = qi[i][j];
    qr[i][j] = a * pr - b * pi;
    qi[i][j] = a * pi + b * pr;
  }
}

__global__ void cg_init_kernel(float* __restrict__ cg) {
  int t = threadIdx.x;
  if (t >= 11) return;
  const int TRI[11][3] = {{0,0,0},{0,1,1},{0,2,2},{1,0,1},{1,1,0},{1,1,2},
                          {1,2,1},{2,0,2},{2,1,1},{2,2,0},{2,2,2}};
  int l1 = TRI[t][0], l2 = TRI[t][1], l3 = TRI[t][2];
  int d1 = 2 * l1 + 1, d2 = 2 * l2 + 1, d3 = 2 * l3 + 1;
  double q1r[5][5], q1i[5][5], q2r[5][5], q2i[5][5], q3r[5][5], q3i[5][5];
  build_q(l1, q1r, q1i); build_q(l2, q2r, q2i); build_q(l3, q3r, q3i);
  double C0[5][5][5];
  for (int i = 0; i < d1; ++i) for (int k = 0; k < d2; ++k) for (int n = 0; n < d3; ++n)
    C0[i][k][n] = su2cg(l1, i - l1, l2, k - l2, l3, n - l3);
  double out[5][5][5]; double nrm = 0.0;
  for (int j = 0; j < d1; ++j) for (int lc = 0; lc < d2; ++lc) for (int m = 0; m < d3; ++m) {
    double sr = 0.0;
    for (int i = 0; i < d1; ++i) for (int k = 0; k < d2; ++k) for (int n = 0; n < d3; ++n) {
      double c0 = C0[i][k][n];
      if (c0 == 0.0) continue;
      double ar = q1r[i][j], ai = q1i[i][j];
      double br = q2r[k][lc], bi = q2i[k][lc];
      double cr = q3r[n][m], ci = -q3i[n][m];
      double abr = ar * br - ai * bi, abi = ar * bi + ai * br;
      sr += (abr * cr - abi * ci) * c0;
    }
    out[j][lc][m] = sr; nrm += sr * sr;
  }
  nrm = sqrt(nrm);
  float* dst = cg + t * 125;
  for (int i = 0; i < 125; ++i) dst[i] = 0.f;
  for (int j = 0; j < d1; ++j) for (int lc = 0; lc < d2; ++lc) for (int m = 0; m < d3; ++m)
    dst[j * 25 + lc * 5 + m] = (float)(out[j][lc][m] / nrm);
}

// ---------------- misc kernels ----------------
__global__ void zero_kernel(float* __restrict__ p, int n) {
  int i = blockIdx.x * blockDim.x + threadIdx.x;
  if (i < n) p[i] = 0.f;
}

__global__ void embed_kernel(const float* __restrict__ x, const float* __restrict__ W,
                             const float* __restrict__ b, float* __restrict__ h) {
  int idx = blockIdx.x * blockDim.x + threadIdx.x;
  if (idx >= NN_NODES * NN_EMB) return;
  int n = idx >> 5, c = idx & 31;
  float s = b[c];
  for (int f = 0; f < NN_FIN; ++f) s += x[n * NN_FIN + f] * W[f * NN_EMB + c];
  h[idx] = s;
}

__global__ void edge_geom_kernel(const float* __restrict__ pos, const int* __restrict__ eidx,
                                 float* __restrict__ sh, float* __restrict__ ef) {
  int e = blockIdx.x * blockDim.x + threadIdx.x;
  if (e >= NN_EDGES) return;
  int s = eidx[e], d = eidx[NN_EDGES + e];
  float vx = pos[s * 3 + 0] - pos[d * 3 + 0];
  float vy = pos[s * 3 + 1] - pos[d * 3 + 1];
  float vz = pos[s * 3 + 2] - pos[d * 3 + 2];
  float len = sqrtf(vx * vx + vy * vy + vz * vz);
  float inv = 1.f / len;
  float X = vx * inv, Y = vy * inv, Z = vz * inv;
  const float s3 = 1.7320508075688772f, s5 = 2.23606797749979f, s15 = 3.872983346207417f;
  float* so = sh + (size_t)e * 9;
  so[0] = 1.f;
  so[1] = s3 * X; so[2] = s3 * Y; so[3] = s3 * Z;
  so[4] = s15 * X * Z;
  so[5] = s15 * X * Y;
  so[6] = s5 * (Y * Y - 0.5f * (X * X + Z * Z));
  so[7] = s15 * Y * Z;
  so[8] = 0.5f * s15 * (Z * Z - X * X);
  float u = len / RMAXF;
  float u5 = u * u * u * u * u;
  float env = 1.f - 21.f * u5 + 35.f * u5 * u - 15.f * u5 * u * u;
  if (len >= RMAXF) env = 0.f;
  float pref = sqrtf(2.f / RMAXF) / len * env;
  const float PI = 3.14159265358979323846f;
  for (int j = 0; j < 8; ++j)
    ef[(size_t)e * 8 + j] = pref * sinf(len * PI * (float)(j + 1) / RMAXF);
}

__global__ void fc1_kernel(const float* __restrict__ ef, const float* __restrict__ W1,
                           const float* __restrict__ b1, unsigned short* __restrict__ hidB) {
  int e = blockIdx.x, m = threadIdx.x;
  __shared__ float sf[8];
  if (m < 8) sf[m] = ef[(size_t)e * 8 + m];
  __syncthreads();
  float s = b1[m];
  for (int j = 0; j < 8; ++j) s += sf[j] * W1[j * NN_MLP + m];
  s = fmaxf(s, 0.f);
  hidB[(size_t)e * NN_MLP + m] = f2bf(s);
}

__global__ void w2cvt_kernel(const float* __restrict__ W2, unsigned short* __restrict__ w2T,
                             int numel) {
  int idx = blockIdx.x * blockDim.x + threadIdx.x;  // one per (n,k), k fastest
  int n = idx >> 8, k = idx & 255;
  if (n >= numel) return;
  w2T[(size_t)n * 256 + k] = f2bf(W2[(size_t)k * numel + n]);
}

// ---------------- fused message kernel: WMMA GEMM + tensor product ----------------
#define TM 16
__global__ __launch_bounds__(256) void msg_layer_kernel(
    const unsigned short* __restrict__ hiddenB,  // [E,256] bf16 (A, row-major)
    const unsigned short* __restrict__ w2T,      // [numel,256] bf16 (B^T, K-contig)
    const float* __restrict__ b2,                // [numel]
    const float* __restrict__ h,                 // [N, in_dim]
    const float* __restrict__ sh,                // [E,9]
    const int* __restrict__ eidx,                // [2,E]
    const float* __restrict__ cg,                // [11,125]
    float* __restrict__ agg,                     // [N,out_dim] zeroed
    LayerCfg cfg) {
  __shared__ __align__(16) unsigned int sA[TM][128];  // 16 edges x 256 bf16 (packed pairs)
  __shared__ float sSh[TM][9];
  __shared__ int   sSrc[TM], sDst[TM];
  __shared__ float sT[TM][32][5];                     // t[e][u][k]
  __shared__ float sW[TM][256];                       // per-u-chunk slice of w
  __shared__ float sMsg[TM][288];

  const int tid = threadIdx.x;
  const int wv = tid >> 5, lane = tid & 31, half = lane >> 4, lanelo = lane & 15;
  const int e0 = blockIdx.x * TM;

  for (int i = tid; i < TM * 32; i += 256) {          // hidden tile via b128 loads
    int e = i >> 5, q = i & 31;
    ((uint4*)&sA[e][0])[q] = ((const uint4*)(hiddenB + (size_t)(e0 + e) * 256))[q];
  }
  for (int i = tid; i < TM * 9; i += 256) sSh[i / 9][i % 9] = sh[(size_t)(e0 + i / 9) * 9 + i % 9];
  if (tid < TM) sSrc[tid] = eidx[e0 + tid];
  if (tid >= TM && tid < 2 * TM) sDst[tid - TM] = eidx[NN_EDGES + e0 + tid - TM];
  for (int i = tid; i < TM * cfg.out_dim; i += 256) sMsg[i / cfg.out_dim][i % cfg.out_dim] = 0.f;
  __syncthreads();

  for (int p = 0; p < cfg.n_paths; ++p) {
    const int l1 = cfg.p_l1[p], l2 = cfg.p_l2[p], l3 = cfg.p_l3[p];
    const int d1 = 2 * l1 + 1, d2 = 2 * l2 + 1, d3 = 2 * l3 + 1;
    const int off1 = cfg.p_i1off[p], shoff = cfg.p_shoff[p], outoff = cfg.p_outoff[p];
    const float* cgv = cg + cfg.p_cg[p] * 125;
    const float pw = cfg.p_pw[p];

    // t[e][u][k] = sum_{i,j} h[dst][off1+u*d1+i] * sh[shoff+j] * CG[i,j,k]
    for (int i = tid; i < TM * 32 * d3; i += 256) {
      int e = i / (32 * d3), rem = i % (32 * d3);
      int u = rem / d3, k = rem % d3;
      const float* hr = h + (size_t)sDst[e] * cfg.in_dim + off1 + u * d1;
      float s = 0.f;
      for (int ii = 0; ii < d1; ++ii) {
        float xv = hr[ii];
        for (int jj = 0; jj < d2; ++jj) s += xv * sSh[e][shoff + jj] * cgv[ii * 25 + jj * 5 + k];
      }
      sT[e][u][k] = s;
    }

    for (int uc = 0; uc < 4; ++uc) {                  // 8 u-values per chunk
      __syncthreads();                                // sT ready / sW free
      const int cbase = cfg.p_coloff[p] + uc * 256;
      // Two 16-col WMMA tiles per wave, computed in ONE K loop with two
      // independent accumulators sharing the A fragment: back-to-back WMMAs
      // have no RAW dependency, and LDS A traffic is halved.
      const int col0 = cbase + (wv * 2 + 0) * 16 + lanelo;
      const int col1 = cbase + (wv * 2 + 1) * 16 + lanelo;
      v8f acc0 = {0.f, 0.f, 0.f, 0.f, 0.f, 0.f, 0.f, 0.f};
      v8f acc1 = {0.f, 0.f, 0.f, 0.f, 0.f, 0.f, 0.f, 0.f};
#pragma unroll
      for (int kc = 0; kc < 8; ++kc) {
        union { unsigned int u[8]; v16bf v; } Af;
        const int M = lanelo;                         // A: M = lane%16 for both halves
#pragma unroll
        for (int r = 0; r < 8; ++r) {
          int kw = 8 * half + r;  // K/2: contiguous k = 16*half + 2r, matching B
          Af.u[r] = sA[M][kc * 16 + kw];
        }
        union { uint4 q[2]; v16bf v; } Bf0, Bf1;      // K-contiguous: two b128 loads each
        const uint4* bp0 = (const uint4*)(w2T + (size_t)col0 * 256 + kc * 32 + 16 * half);
        const uint4* bp1 = (const uint4*)(w2T + (size_t)col1 * 256 + kc * 32 + 16 * half);
        Bf0.q[0] = bp0[0]; Bf0.q[1] = bp0[1];
        Bf1.q[0] = bp1[0]; Bf1.q[1] = bp1[1];
        __builtin_prefetch(bp0 + 4, 0, 1);            // global_prefetch next K-slab
        acc0 = __builtin_amdgcn_wmma_f32_16x16x32_bf16(false, Af.v, false, Bf0.v,
                                                       (short)0, acc0, false, false);
        acc1 = __builtin_amdgcn_wmma_f32_16x16x32_bf16(false, Af.v, false, Bf1.v,
                                                       (short)0, acc1, false, false);
      }
      const float bias0 = b2[col0];
      const float bias1 = b2[col1];
#pragma unroll
      for (int r = 0; r < 8; ++r) {
        sW[r + 8 * half][(wv * 2 + 0) * 16 + lanelo] = acc0[r] + bias0;
        sW[r + 8 * half][(wv * 2 + 1) * 16 + lanelo] = acc1[r] + bias1;
      }
      __syncthreads();
      // out[e][w][k] += pw * sum_u t[e][u][k] * W[e][u][w]
      for (int i = tid; i < TM * 32; i += 256) {
        int e = i >> 5, w = i & 31;
        for (int k = 0; k < d3; ++k) {
          float s = 0.f;
#pragma unroll
          for (int du = 0; du < 8; ++du) s += sT[e][uc * 8 + du][k] * sW[e][du * 32 + w];
          sMsg[e][outoff + w * d3 + k] += pw * s;
        }
      }
    }
    __syncthreads();  // before next path rewrites sT
  }

  for (int i = tid; i < TM * cfg.out_dim; i += 256) {
    int e = i / cfg.out_dim, c = i % cfg.out_dim;
    atomicAdd(&agg[(size_t)sSrc[e] * cfg.out_dim + c], sMsg[e][c]);
  }
}

// ---------------- batchnorm ----------------
__global__ void bn_stats_kernel(const float* __restrict__ agg, const float* __restrict__ bnw,
                                float* __restrict__ stats, LayerCfg cfg) {
  int ch = blockIdx.x, f = ch / 32, m = ch % 32;
  int l = cfg.out_l[f], d = 2 * l + 1;
  int off = cfg.out_off[f] + m * d;
  int tid = threadIdx.x;
  __shared__ float red[256];
  float s = 0.f;
  if (l == 0)
    for (int n = tid; n < NN_NODES; n += 256) s += agg[(size_t)n * cfg.out_dim + off];
  red[tid] = s; __syncthreads();
  for (int st = 128; st; st >>= 1) { if (tid < st) red[tid] += red[tid + st]; __syncthreads(); }
  float mean = (l == 0) ? red[0] / (float)NN_NODES : 0.f;
  __syncthreads();
  s = 0.f;
  for (int n = tid; n < NN_NODES; n += 256)
    for (int k = 0; k < d; ++k) {
      float v = agg[(size_t)n * cfg.out_dim + off + k] - mean;
      s += v * v;
    }
  red[tid] = s; __syncthreads();
  for (int st = 128; st; st >>= 1) { if (tid < st) red[tid] += red[tid + st]; __syncthreads(); }
  if (tid == 0) {
    float norm = red[0] / (float)(NN_NODES * d);
    stats[ch * 2] = mean;
    stats[ch * 2 + 1] = bnw[ch] / sqrtf(norm + 1e-5f);
  }
}

__global__ void bn_apply_kernel(const float* __restrict__ agg, const float* __restrict__ stats,
                                const float* __restrict__ bnb, const float* __restrict__ h_old,
                                float* __restrict__ h_new, LayerCfg cfg) {
  int idx = blockIdx.x * blockDim.x + threadIdx.x;
  if (idx >= NN_NODES * cfg.out_dim) return;
  int n = idx / cfg.out_dim, c = idx % cfg.out_dim;
  int f = 0;
  while (f + 1 < cfg.n_out_fields && c >= cfg.out_off[f + 1]) ++f;
  int l = cfg.out_l[f], d = 2 * l + 1;
  int m = (c - cfg.out_off[f]) / d;
  int ch = f * 32 + m;
  float v = agg[idx];
  float mean = stats[ch * 2], scale = stats[ch * 2 + 1];
  if (l == 0) v = (v - mean) * scale + bnb[m];
  else v = v * scale;
  float res = (c < cfg.in_dim) ? h_old[(size_t)n * cfg.in_dim + c] : 0.f;
  h_new[idx] = v + res;
}

__global__ void finalize_kernel(const float* __restrict__ h, const int* __restrict__ batch,
                                float* __restrict__ out) {
  int idx = blockIdx.x * blockDim.x + threadIdx.x;
  if (idx >= NN_NODES * NN_EMB) return;
  int n = idx / NN_EMB, c = idx % NN_EMB;
  float v = h[idx];
  out[idx] = v;
  atomicAdd(&out[NN_NODES * NN_EMB + batch[n] * NN_EMB + c], v);
}

// ---------------- host ----------------
static void build_cfgs(LayerCfg* cfgs) {
  const int O[3] = {0, 32, 128}, S_[3] = {0, 1, 4};
  {
    LayerCfg& c = cfgs[0];
    c.numel = 3072; c.in_dim = 32; c.out_dim = 288; c.n_paths = 3; c.n_out_fields = 3;
    for (int f = 0; f < 3; ++f) { c.out_off[f] = O[f]; c.out_l[f] = f; }
    for (int p = 0; p < 3; ++p) {
      c.p_i1off[p] = 0; c.p_l1[p] = 0;
      c.p_shoff[p] = S_[p]; c.p_l2[p] = p;
      c.p_outoff[p] = O[p]; c.p_l3[p] = p;
      c.p_cg[p] = p; c.p_coloff[p] = p * 1024;
      c.p_pw[p] = sqrtf((2.f * p + 1.f) / 32.f);
    }
  }
  const int P[11][4] = {{0,0,0,0},{0,1,1,1},{0,2,2,2},{1,0,1,3},{1,1,0,4},{1,1,2,5},
                        {1,2,1,6},{2,0,2,7},{2,1,1,8},{2,2,0,9},{2,2,2,10}};
  const float PW[3] = {sqrtf(1.f / 96.f), sqrtf(3.f / 128.f), sqrtf(5.f / 128.f)};
  for (int li = 1; li <= 2; ++li) {
    LayerCfg& c = cfgs[li];
    c.numel = 11264; c.in_dim = 288; c.out_dim = 288; c.n_paths = 11; c.n_out_fields = 3;
    for (int f = 0; f < 3; ++f) { c.out_off[f] = O[f]; c.out_l[f] = f; }
    for (int p = 0; p < 11; ++p) {
      int i1 = P[p][0], i2 = P[p][1], i3 = P[p][2];
      c.p_i1off[p] = O[i1]; c.p_l1[p] = i1;
      c.p_shoff[p] = S_[i2]; c.p_l2[p] = i2;
      c.p_outoff[p] = O[i3]; c.p_l3[p] = i3;
      c.p_cg[p] = P[p][3]; c.p_coloff[p] = p * 1024;
      c.p_pw[p] = PW[i3];
    }
  }
  {
    LayerCfg& c = cfgs[3];
    c.numel = 3072; c.in_dim = 288; c.out_dim = 32; c.n_paths = 3; c.n_out_fields = 1;
    c.out_off[0] = 0; c.out_l[0] = 0;
    const int P3[3][2] = {{0, 0}, {1, 4}, {2, 9}};
    for (int p = 0; p < 3; ++p) {
      int i = P3[p][0];
      c.p_i1off[p] = O[i]; c.p_l1[p] = i;
      c.p_shoff[p] = S_[i]; c.p_l2[p] = i;
      c.p_outoff[p] = 0; c.p_l3[p] = 0;
      c.p_cg[p] = P3[p][1]; c.p_coloff[p] = p * 1024;
      c.p_pw[p] = sqrtf(1.f / 96.f);
    }
  }
}

extern "C" void kernel_launch(void* const* d_in, const int* in_sizes, int n_in,
                              void* d_out, int out_size, void* d_ws, size_t ws_size,
                              hipStream_t stream) {
  (void)n_in; (void)out_size; (void)ws_size;
  const float* X    = (const float*)d_in[0];
  const float* POS  = (const float*)d_in[1];
  const float* EMBW = (const float*)d_in[2];
  const float* EMBB = (const float*)d_in[3];
  struct LP { const float *w1, *b1, *w2, *b2, *bnw, *bnb; };
  LP lp[4];
  for (int li = 0; li < 4; ++li) {
    int g = 4 + li * 6;
    if (in_sizes[g] == 2048) {  // insertion order: fc1_W, fc1_b, fc2_W, fc2_b, bn_w, bn_b
      lp[li] = {(const float*)d_in[g],     (const float*)d_in[g + 1],
                (const float*)d_in[g + 2], (const float*)d_in[g + 3],
                (const float*)d_in[g + 4], (const float*)d_in[g + 5]};
    } else {  // jax-sorted: bn_b, bn_w, fc1_W, fc1_b, fc2_W, fc2_b
      lp[li] = {(const float*)d_in[g + 2], (const float*)d_in[g + 3],
                (const float*)d_in[g + 4], (const float*)d_in[g + 5],
                (const float*)d_in[g + 1], (const float*)d_in[g]};
    }
  }
  const int* EIDX  = (const int*)d_in[28];
  const int* BATCH = (const int*)d_in[29];
  float* OUT = (float*)d_out;

  char* ws = (char*)d_ws;
  size_t off = 0;
  auto take = [&](size_t bytes) -> void* {
    off = (off + 255) & ~(size_t)255;
    void* p = ws + off; off += bytes; return p;
  };
  float* cg  = (float*)take(11 * 125 * 4);
  float* hA  = (float*)take((size_t)NN_NODES * 288 * 4);
  float* hB  = (float*)take((size_t)NN_NODES * 288 * 4);
  float* agg = (float*)take((size_t)NN_NODES * 288 * 4);
  float* shb = (float*)take((size_t)NN_EDGES * 9 * 4);
  float* efb = (float*)take((size_t)NN_EDGES * 8 * 4);
  unsigned short* hidB = (unsigned short*)take((size_t)NN_EDGES * 256 * 2);
  unsigned short* w2T  = (unsigned short*)take((size_t)256 * 11264 * 2);
  float* stats = (float*)take(96 * 2 * 4);

  LayerCfg cfgs[4];
  build_cfgs(cfgs);

  cg_init_kernel<<<1, 32, 0, stream>>>(cg);
  embed_kernel<<<(NN_NODES * NN_EMB + 255) / 256, 256, 0, stream>>>(X, EMBW, EMBB, hA);
  edge_geom_kernel<<<(NN_EDGES + 255) / 256, 256, 0, stream>>>(POS, EIDX, shb, efb);

  for (int li = 0; li < 4; ++li) {
    const LayerCfg& c = cfgs[li];
    fc1_kernel<<<NN_EDGES, 256, 0, stream>>>(efb, lp[li].w1, lp[li].b1, hidB);
    w2cvt_kernel<<<c.numel, 256, 0, stream>>>(lp[li].w2, w2T, c.numel);
    zero_kernel<<<(NN_NODES * c.out_dim + 255) / 256, 256, 0, stream>>>(agg, NN_NODES * c.out_dim);
    msg_layer_kernel<<<NN_EDGES / TM, 256, 0, stream>>>(hidB, w2T, lp[li].b2, hA, shb, EIDX,
                                                        cg, agg, c);
    bn_stats_kernel<<<32 * c.n_out_fields, 256, 0, stream>>>(agg, lp[li].bnw, stats, c);
    bn_apply_kernel<<<(NN_NODES * c.out_dim + 255) / 256, 256, 0, stream>>>(
        agg, stats, lp[li].bnb, hA, hB, c);
    float* t = hA; hA = hB; hB = t;
  }

  zero_kernel<<<1, 256, 0, stream>>>(OUT + NN_NODES * NN_EMB, NN_GRAPHS * NN_EMB);
  finalize_kernel<<<(NN_NODES * NN_EMB + 255) / 256, 256, 0, stream>>>(hA, BATCH, OUT);
}
